// MultiHeadAttention_20401094656661
// MI455X (gfx1250) — matrix operand order repaired and checked by reading, not compiled, not run on hardware
//
#include <hip/hip_runtime.h>
#include <hip/hip_bf16.h>

typedef __bf16 bf16;
typedef __attribute__((ext_vector_type(8)))  __bf16 v8bf;
typedef __attribute__((ext_vector_type(16))) __bf16 v16bf;
typedef __attribute__((ext_vector_type(8)))  float  v8f;
typedef __attribute__((ext_vector_type(4)))  unsigned int u32x4;
typedef __attribute__((ext_vector_type(8)))  int i32x8;
typedef __attribute__((ext_vector_type(4)))  int i32x4;

#if __has_builtin(__builtin_amdgcn_tensor_load_to_lds) && __has_builtin(__builtin_amdgcn_s_wait_tensorcnt)
#define HAS_TDM 1
#else
#define HAS_TDM 0
#endif

static __device__ __forceinline__ v16bf cat16(v8bf lo, v8bf hi) {
  return __builtin_shufflevector(lo, hi, 0,1,2,3,4,5,6,7,8,9,10,11,12,13,14,15);
}
static __device__ __forceinline__ v8f wmma_bf16(v16bf a, v16bf b, v8f c) {
  return __builtin_amdgcn_wmma_f32_16x16x32_bf16(false, a, false, b, (short)0, c, false, false);
}
// A-frag (16x32, MxK): lane=row(l15); K chunks {hl*8..+7} and {16+hl*8..+7}
static __device__ __forceinline__ v16bf load_afrag(const bf16* rowBase, int hl) {
  return cat16(*(const v8bf*)(rowBase + hl * 8), *(const v8bf*)(rowBase + hl * 8 + 16));
}
// B-frag (32x16, KxN): lane=col(l15); K chunks {hl*8..+7} and {16+hl*8..+7}
static __device__ __forceinline__ v16bf load_bfrag(const bf16* colBase, int hl) {
  return cat16(*(const v8bf*)(colBase + hl * 8), *(const v8bf*)(colBase + hl * 8 + 16));
}

static __device__ __forceinline__ unsigned int ldsoff(const void* p) {
  return (unsigned int)(size_t)p;   // LDS aperture: low 32 bits = LDS byte offset
}

#if HAS_TDM
// 2-D TDM load: 64x64 bf16 tile, row stride `strideElems`, into LDS at ldsOff.
static __device__ __forceinline__ void tdm_load_tile64x64_bf16(
    unsigned int ldsOff, const bf16* g, int strideElems) {
  unsigned long long ga = (unsigned long long)(size_t)g;
  u32x4 g0;
  g0[0] = 1u;                                   // count=1, user descriptor
  g0[1] = ldsOff;                               // lds_addr
  g0[2] = (unsigned int)ga;                     // global_addr[31:0]
  g0[3] = (unsigned int)((ga >> 32) & 0x01FFFFFFu) | 0x80000000u; // addr[56:32] | type=2
  i32x8 g1;
  g1[0] = 0x00010000;                           // workgroup_mask=0, data_size=2B
  g1[1] = (int)(((unsigned)strideElems & 0xFFFFu) << 16);        // tensor_dim0 lo16
  g1[2] = (int)((((unsigned)strideElems) >> 16) | 0xFFFF0000u);  // dim0 hi | dim1 lo=0xFFFF
  g1[3] = (64 << 16);                           // dim1 hi=0 | tile_dim0=64
  g1[4] = 64;                                   // tile_dim1=64, tile_dim2=0
  g1[5] = strideElems;                          // tensor_dim0_stride lo32
  g1[6] = 0;
  g1[7] = 0;
  i32x4 z4 = {0, 0, 0, 0};
#if defined(__clang_major__) && (__clang_major__ >= 23)
  i32x8 z8 = {0, 0, 0, 0, 0, 0, 0, 0};
  __builtin_amdgcn_tensor_load_to_lds(g0, g1, z4, z4, z8, 0);
#else
  __builtin_amdgcn_tensor_load_to_lds(g0, g1, z4, z4, 0);
#endif
}
#endif

// Fallback staging: 64x64 bf16 tile copy, 128 threads.
static __device__ __forceinline__ void stage_tile64(bf16* dst, const bf16* src,
                                                    int strideElems, int tid) {
#pragma unroll
  for (int i = 0; i < 4; ++i) {
    int c = tid + i * 128;            // 512 chunks of 8 bf16
    int r = c >> 3, c8 = c & 7;
    *(v8bf*)(dst + r * 64 + c8 * 8) =
        *(const v8bf*)(src + (size_t)r * strideElems + c8 * 8);
  }
}

// ---------------------------------------------------------------------------
// C[M,N] = A[M,K] * W[K,N] + bias, out = (acc+bias)*outScale.
// IN_BF16: A is bf16, else fp32.  OUT_BF16: C is bf16, else fp32.
// ---------------------------------------------------------------------------
template <int IN_BF16, int OUT_BF16>
__global__ __launch_bounds__(256) void gemm_bias_wmma(
    const void* __restrict__ Av, const float* __restrict__ W,
    const float* __restrict__ bias, void* __restrict__ Cv,
    int M, int N, int K, float outScale) {
  __shared__ bf16 As[128 * 32];   // [m][k]
  __shared__ bf16 Ws[64 * 32];    // [n][k] (transposed on load)

  const int tid = threadIdx.x, lane = tid & 31, wave = tid >> 5;
  const int hl = lane >> 4, l15 = lane & 15;
  const int wm = wave >> 1, wn = wave & 1;
  const int mBase = blockIdx.x * 128, nBase = blockIdx.y * 64;

  v8f acc[2][2] = {};

  for (int k0 = 0; k0 < K; k0 += 32) {
    float4 afv[4]; v8bf abv[2]; float4 wfv[2];
    if (IN_BF16) {
      const bf16* Ag = (const bf16*)Av + (size_t)mBase * K + k0;
#pragma unroll
      for (int it = 0; it < 2; ++it) {
        int idx = tid + it * 256;
        int r = idx >> 2, c8 = idx & 3;
        abv[it] = *(const v8bf*)(Ag + (size_t)r * K + c8 * 8);
      }
    } else {
      const float* Ag = (const float*)Av + (size_t)mBase * K + k0;
#pragma unroll
      for (int it = 0; it < 4; ++it) {
        int idx = tid + it * 256;
        int r = idx >> 3, c4 = idx & 7;
        afv[it] = *(const float4*)(Ag + (size_t)r * K + c4 * 4);
      }
      if (k0 + 32 < K)
        __builtin_prefetch(Ag + (size_t)(tid >> 1) * K + 32, 0, 1);
    }
    const float* Wg = W + (size_t)k0 * N + nBase;
#pragma unroll
    for (int it = 0; it < 2; ++it) {
      int idx = tid + it * 256;
      int r = idx >> 4, c4 = idx & 15;
      wfv[it] = *(const float4*)(Wg + (size_t)r * N + c4 * 4);
    }
    if (IN_BF16) {
#pragma unroll
      for (int it = 0; it < 2; ++it) {
        int idx = tid + it * 256;
        int r = idx >> 2, c8 = idx & 3;
        *(v8bf*)(As + r * 32 + c8 * 8) = abv[it];
      }
    } else {
#pragma unroll
      for (int it = 0; it < 4; ++it) {
        int idx = tid + it * 256;
        int r = idx >> 3, c4 = idx & 7;
        bf16* d = As + r * 32 + c4 * 4;
        d[0] = (bf16)afv[it].x; d[1] = (bf16)afv[it].y;
        d[2] = (bf16)afv[it].z; d[3] = (bf16)afv[it].w;
      }
    }
#pragma unroll
    for (int it = 0; it < 2; ++it) {
      int idx = tid + it * 256;
      int r = idx >> 4, c4 = idx & 15;
      int n = c4 * 4;
      Ws[(n + 0) * 32 + r] = (bf16)wfv[it].x;
      Ws[(n + 1) * 32 + r] = (bf16)wfv[it].y;
      Ws[(n + 2) * 32 + r] = (bf16)wfv[it].z;
      Ws[(n + 3) * 32 + r] = (bf16)wfv[it].w;
    }
    __syncthreads();

    v16bf af[2], bfr[2];
#pragma unroll
    for (int i = 0; i < 2; ++i) {
      af[i]  = load_afrag(As + (wm * 32 + i * 16 + l15) * 32, hl);
      bfr[i] = load_bfrag(Ws + (wn * 32 + i * 16 + l15) * 32, hl);
    }
#pragma unroll
    for (int i = 0; i < 2; ++i)
#pragma unroll
      for (int j = 0; j < 2; ++j)
        acc[i][j] = wmma_bf16(af[i], bfr[j], acc[i][j]);
    __syncthreads();
  }

#pragma unroll
  for (int j = 0; j < 2; ++j) {
    int col = nBase + wn * 32 + j * 16 + l15;
    float b = bias[col];
#pragma unroll
    for (int i = 0; i < 2; ++i)
#pragma unroll
      for (int r = 0; r < 8; ++r) {
        int row = mBase + wm * 32 + i * 16 + r + hl * 8;
        float val = (acc[i][j][r] + b) * outScale;
        if (OUT_BF16) ((bf16*)Cv)[(size_t)row * N + col] = (bf16)val;
        else          ((float*)Cv)[(size_t)row * N + col] = val;
      }
  }
}

// ---------------------------------------------------------------------------
// Flash attention over bf16 qp/vp (Q pre-scaled by 1/8; K==V==vp per ref bug).
// Transposed-score scheme: per wave, S^T = K_tile x Q_wave^T, so
//   C-frag: lane = q column, VGPR = key row  ->
//   * softmax stats per-lane scalars (one shfl_xor(16) per reduction)
//   * exp'd fragments ARE the P A-frags for P*V (no LDS round-trip)
// ---------------------------------------------------------------------------
__global__ __launch_bounds__(128) void flash_attn_wmma(
    const bf16* __restrict__ QP, const bf16* __restrict__ VP,
    bf16* __restrict__ CTX, int S) {
  const int Dm = 1024;
  const int qT = blockIdx.x, h = blockIdx.y, b = blockIdx.z;
  const int tid = threadIdx.x, lane = tid & 31, wave = tid >> 5;
  const int hl = lane >> 4, l15 = lane & 15;

  __shared__ bf16 Kbuf[2][64 * 64];   // [key][d], TDM double buffer
  __shared__ bf16 Vt[64 * 64];        // Q staging, then [d][key] transpose

  const bf16* qTile = QP + ((size_t)b * S + qT * 64) * Dm + h * 64;
  const bf16* vBase = VP + (size_t)b * S * Dm + h * 64;
  bf16*       oBase = CTX + ((size_t)b * S + qT * 64) * Dm + h * 64;
  const int NT = S / 64;

#if HAS_TDM
  if (wave == 0) {
    tdm_load_tile64x64_bf16(ldsoff(Vt), qTile, Dm);
    tdm_load_tile64x64_bf16(ldsoff(Kbuf[0]), vBase, Dm);
    __builtin_amdgcn_s_wait_tensorcnt(1);   // Q tile done (in-order)
  }
#else
  stage_tile64(Vt, qTile, Dm, tid);
  stage_tile64(Kbuf[0], vBase, Dm, tid);
#endif
  __syncthreads();

  // B-frags of Q^T for this wave's 16 q columns (K-dim = d, contiguous rows)
  v16bf bq[2];
#pragma unroll
  for (int c = 0; c < 2; ++c)
    bq[c] = load_bfrag(Vt + (wave * 16 + l15) * 64 + c * 32, hl);

  v8f accO[4] = {};
  float m = -1e30f, l = 0.f;      // per-lane stats for q column l15 of wave tile

  for (int kt = 0; kt < NT; ++kt) {
#if HAS_TDM
    if (wave == 0) {
      if (kt + 1 < NT) {
        tdm_load_tile64x64_bf16(ldsoff(Kbuf[(kt + 1) & 1]),
                                vBase + (size_t)(kt + 1) * 64 * Dm, Dm);
        __builtin_amdgcn_s_wait_tensorcnt(1);   // tile kt complete
      } else {
        __builtin_amdgcn_s_wait_tensorcnt(0);
      }
    }
#else
    if (kt > 0) stage_tile64(Kbuf[kt & 1], vBase + (size_t)kt * 64 * Dm, Dm, tid);
#endif
    __syncthreads();                  // B1: K tile kt visible; prior reads done
    const bf16* Kl = Kbuf[kt & 1];

    // ---- build Vt = K tile transposed (LDS->LDS) ----
#pragma unroll
    for (int i = 0; i < 4; ++i) {
      int c = tid + i * 128;
      int r = c >> 3, c8 = c & 7;
      v8bf x = *(const v8bf*)(Kl + r * 64 + c8 * 8);
#pragma unroll
      for (int j = 0; j < 8; ++j) Vt[(c8 * 8 + j) * 64 + r] = x[j];
    }

    // ---- S^T frags: st[mt] = 16 keys x 16 q (keys mt*16 + r + 8*hl) ----
    v8f st[4];
#pragma unroll
    for (int mt = 0; mt < 4; ++mt) {
      const bf16* kr = Kl + (mt * 16 + l15) * 64;
      v8f z = {};
      z = wmma_bf16(load_afrag(kr, hl), bq[0], z);
      z = wmma_bf16(load_afrag(kr + 32, hl), bq[1], z);
      st[mt] = z;
    }

    // ---- per-lane online softmax (q = wave*16 + l15) ----
    float mx = -1e30f;
#pragma unroll
    for (int mt = 0; mt < 4; ++mt)
#pragma unroll
      for (int r = 0; r < 8; ++r) mx = fmaxf(mx, st[mt][r]);
    mx = fmaxf(mx, __shfl_xor(mx, 16, 32));     // join the two key halves
    float mnew = fmaxf(m, mx);
    float s = 0.f;
#pragma unroll
    for (int mt = 0; mt < 4; ++mt)
#pragma unroll
      for (int r = 0; r < 8; ++r) {
        float p = __expf(st[mt][r] - mnew);
        st[mt][r] = p;
        s += p;
      }
    s += __shfl_xor(s, 16, 32);
    float cf = __expf(m - mnew);
    l = l * cf + s;
    m = mnew;

    // ---- pack P into A-frags, purely in-register ----
    v16bf ap0{}, ap1{};
#pragma unroll
    for (int r = 0; r < 8; ++r) {
      ap0[r]     = (bf16)st[0][r];
      ap0[r + 8] = (bf16)st[1][r];
      ap1[r]     = (bf16)st[2][r];
      ap1[r + 8] = (bf16)st[3][r];
    }

    // ---- rescale accO: cf for q-row (r + 8*hl) lives in lane (r + 8*hl) ----
#pragma unroll
    for (int r = 0; r < 8; ++r) {
      float cfr = __shfl(cf, r + 8 * hl, 32);
#pragma unroll
      for (int dt = 0; dt < 4; ++dt) accO[dt][r] *= cfr;
    }
    __syncthreads();                  // B2: Vt transpose complete

    // ---- O += P * V ----
#pragma unroll
    for (int dt = 0; dt < 4; ++dt) {
      const bf16* vr = Vt + (dt * 16 + l15) * 64;
      accO[dt] = wmma_bf16(ap0, load_bfrag(vr, hl), accO[dt]);
      accO[dt] = wmma_bf16(ap1, load_bfrag(vr + 32, hl), accO[dt]);
    }
  }

  // ---- finalize: divide row q by l_q (broadcast from lane q), store bf16 ----
#pragma unroll
  for (int r = 0; r < 8; ++r) {
    float lr = __shfl(l, r + 8 * hl, 32);
    float inv = 1.f / lr;
    int row = wave * 16 + r + 8 * hl;
#pragma unroll
    for (int dt = 0; dt < 4; ++dt)
      oBase[(size_t)row * Dm + dt * 16 + l15] = (bf16)(accO[dt][r] * inv);
  }
}

// ---------------------------------------------------------------------------
extern "C" void kernel_launch(void* const* d_in, const int* in_sizes, int n_in,
                              void* d_out, int out_size, void* d_ws, size_t ws_size,
                              hipStream_t stream) {
  (void)in_sizes; (void)n_in; (void)out_size; (void)ws_size;
  const float* q  = (const float*)d_in[0];
  // d_in[1]=k, d_in[5]=Wk, d_in[6]=bk: dead (reference bug — kp unused)
  const float* v  = (const float*)d_in[2];
  const float* Wq = (const float*)d_in[3];
  const float* bq = (const float*)d_in[4];
  const float* Wv = (const float*)d_in[7];
  const float* bv = (const float*)d_in[8];
  const float* Wo = (const float*)d_in[9];
  const float* bo = (const float*)d_in[10];
  float* out = (float*)d_out;

  const int B = 2, S = 2048, D = 1024, M = B * S;
  bf16* qp  = (bf16*)d_ws;                 // 8 MB, pre-scaled by 1/8
  bf16* vp  = qp + (size_t)M * D;          // 8 MB
  bf16* ctx = vp + (size_t)M * D;          // 8 MB

  dim3 gblk(256);
  dim3 ggrid(M / 128, D / 64);
  gemm_bias_wmma<0, 1><<<ggrid, gblk, 0, stream>>>(q, Wq, bq, qp, M, D, D, 0.125f);
  gemm_bias_wmma<0, 1><<<ggrid, gblk, 0, stream>>>(v, Wv, bv, vp, M, D, D, 1.0f);

  dim3 agrid(S / 64, 16, B);
  flash_attn_wmma<<<agrid, dim3(128), 0, stream>>>(qp, vp, ctx, S);

  gemm_bias_wmma<1, 0><<<ggrid, gblk, 0, stream>>>(ctx, Wo, bo, out, M, D, D, 1.0f);
}
